// LocallyConnectedLayer_18734647345342
// MI455X (gfx1250) — matrix operand order repaired and checked by reading, not compiled, not run on hardware
//
#include <hip/hip_runtime.h>

typedef __attribute__((ext_vector_type(16))) __bf16       v16bf;
typedef __attribute__((ext_vector_type(8)))  float        v8f;
typedef __attribute__((ext_vector_type(4)))  float        v4f;
typedef __attribute__((ext_vector_type(4)))  unsigned int v4u;
typedef __attribute__((ext_vector_type(8)))  int          v8i;
typedef __attribute__((ext_vector_type(4)))  int          v4i;

#define BATCH  32
#define CIN    96
#define HH     32
#define WW     32
#define OC     96
#define OH     30
#define OW     30
#define KDIM   864      // CIN * 3 * 3
#define NPATCH 900      // OH * OW

// LDS row stride in dwords after TDM padding: 864 data dwords + 27 pads of 4 dwords.
// 972 mod 64 = 12 -> 16 rows hit 16 distinct bank groups, conflict-free b128 reads.
#define ROWDW  972

__global__ __launch_bounds__(384) void lc_wmma_kernel(
    const float* __restrict__ x, const float* __restrict__ wgt,
    const float* __restrict__ bias, float* __restrict__ out)
{
  __shared__ unsigned int smem[BATCH * ROWDW];   // 124,416 B: A[b][k] fp32, TDM-padded

  const int p  = blockIdx.x;
  const int pi = p / OW;
  const int pj = p - pi * OW;

  const int tid  = threadIdx.x;
  const int wave = tid >> 5;
  const int lane = tid & 31;

  // ---- Phase 1: TDM im2col -> LDS (wave 0 issues one 4-D tile DMA) ----
  if (wave == 0) {
    unsigned long long gaddr =
        (unsigned long long)(const void*)(x + (size_t)(pi * WW + pj));
    unsigned int ldsa = (unsigned int)(unsigned long long)(void*)&smem[0];

    v4u g0;
    g0[0] = 1u;                                   // count=1, user descriptor
    g0[1] = ldsa;                                 // lds_addr (bytes)
    g0[2] = (unsigned int)gaddr;                  // global_addr[31:0]
    g0[3] = (unsigned int)((gaddr >> 32) & 0x1FFFFFFull) | (2u << 30); // addr[56:32] | type=2

    v8i g1;
    // data_size=4B (2<<16), pad_enable (1<<20), pad every 32 dwords (code 4<<22),
    // pad amount 4 dwords (code 3<<25); workgroup_mask=0 (not in a cluster)
    g1[0] = (2 << 16) | (1 << 20) | (4 << 22) | (3 << 25);
    g1[1] = 0;                                    // atomic_barrier_addr=0; tensor_dim0 lo16 (1<<20 -> 0)
    g1[2] = 16;                                   // tensor_dim0 hi16 = 16; tensor_dim1 lo16 = 0
    g1[3] = 16 | (3 << 16);                       // tensor_dim1 hi16 = 16; tile_dim0 = 3 (dj)
    g1[4] = 3 | (96 << 16);                       // tile_dim1 = 3 (di); tile_dim2 = 96 (c)
    g1[5] = WW;                                   // tensor_dim0_stride = W (rows)
    g1[6] = (int)((unsigned)(HH * WW) << 16);     // stride0 hi16=0; stride1 lo16 = H*W (channels)
    g1[7] = 0;                                    // stride1 hi bits

    v4i g2;
    g2[0] = 1 << 20;                              // tensor_dim2 (large: no clipping)
    g2[1] = 1 << 20;                              // tensor_dim3 (large: no clipping)
    g2[2] = CIN * HH * WW;                        // tensor_dim2_stride = C*H*W (batch)
    g2[3] = (BATCH << 16);                        // stride2 hi16=0; tile_dim3 = 32 (b)

    v4i g3 = {0, 0, 0, 0};                        // no dim4
    v8i g4 = {0, 0, 0, 0, 0, 0, 0, 0};            // extra operand (clang-23 form), zero-filled

    __builtin_amdgcn_tensor_load_to_lds(g0, g1, g2, g3, g4, 0);
    __builtin_amdgcn_s_wait_tensorcnt(0);
  }
  __syncthreads();

  // ---- Phase 2: per-wave 16x16 tile, K-loop of bf16 WMMA ----
  const int m0   = (wave & 1) << 4;               // batch tile base
  const int n0   = (wave >> 1) << 4;              // out-channel tile base
  const int half = lane >> 4;                     // lane half selects K sub-block
  const int lm   = lane & 15;

  const int brow = m0 + lm;                       // A row (batch index)
  const int orow = n0 + lm;                       // B row (out channel)
  const float* wrow = wgt + ((size_t)p * OC + orow) * KDIM + half * 16;
  const unsigned int abase = (unsigned int)brow * ROWDW;
  const int ka = half * 16;

  v8f acc = {0.f, 0.f, 0.f, 0.f, 0.f, 0.f, 0.f, 0.f};

  for (int kc = 0; kc < KDIM; kc += 32) {
    // A: 16 consecutive k (k = 16*half + s, matching B), conflict-free b128 LDS reads
    const int k0 = kc + ka;
    const int k1 = k0 + 8;
    const v4f* ap0 = (const v4f*)&smem[abase + k0 + ((k0 >> 5) << 2)];
    const v4f* ap1 = (const v4f*)&smem[abase + k1 + ((k1 >> 5) << 2)];
    v4f a0 = ap0[0], a1 = ap0[1];
    v4f a2 = ap1[0], a3 = ap1[1];

    // B: 16 consecutive fp32 weights, 64B contiguous, streamed nontemporally
    const v4f* wp = (const v4f*)(wrow + kc);
    v4f w0 = __builtin_nontemporal_load(wp + 0);
    v4f w1 = __builtin_nontemporal_load(wp + 1);
    v4f w2 = __builtin_nontemporal_load(wp + 2);
    v4f w3 = __builtin_nontemporal_load(wp + 3);

    v16bf A, B;
#pragma unroll
    for (int i = 0; i < 4; ++i) {
      A[i]      = (__bf16)a0[i];
      A[4 + i]  = (__bf16)a1[i];
      A[8 + i]  = (__bf16)a2[i];
      A[12 + i] = (__bf16)a3[i];
      B[i]      = (__bf16)w0[i];
      B[4 + i]  = (__bf16)w1[i];
      B[8 + i]  = (__bf16)w2[i];
      B[12 + i] = (__bf16)w3[i];
    }

    acc = __builtin_amdgcn_wmma_f32_16x16x32_bf16(
        /*neg_a=*/false, A, /*neg_b=*/false, B,
        /*c_mod=*/(short)0, acc, /*reuse_a=*/false, /*reuse_b=*/false);
  }

  // ---- Epilogue: bias + store in [B, OC, OH, OW] layout ----
  const float bv = bias[(size_t)p * OC + orow];
#pragma unroll
  for (int r = 0; r < 8; ++r) {
    const int bb = m0 + (half << 3) + r;          // C/D layout: M = r (+8 for hi lanes)
    out[((size_t)bb * OC + orow) * (OH * OW) + p] = acc[r] + bv;
  }
}

extern "C" void kernel_launch(void* const* d_in, const int* in_sizes, int n_in,
                              void* d_out, int out_size, void* d_ws, size_t ws_size,
                              hipStream_t stream) {
  (void)in_sizes; (void)n_in; (void)d_ws; (void)ws_size; (void)out_size;
  const float* x    = (const float*)d_in[0];
  const float* wgt  = (const float*)d_in[1];
  const float* bias = (const float*)d_in[2];
  float* out        = (float*)d_out;
  lc_wmma_kernel<<<NPATCH, 384, 0, stream>>>(x, wgt, bias, out);
}
